// DilatedSelfAttention_4879082848627
// MI455X (gfx1250) — matrix operand order repaired and checked by reading, not compiled, not run on hardware
//
#include <hip/hip_runtime.h>

// ---------------------------------------------------------------- problem dims
#define DIM   1024
#define HEADS 16
#define DK    64
#define SEG   512
#define NNE   2
#define DIL   4
#define BB    2
#define LL    8192
#define SS    (LL / SEG)      // 16 segments
#define M_TOT (BB * LL)       // 16384 rows

// GEMM tiling
#define TM  128
#define TN  128
#define TKG 64                // k-slab per staging round (2 WMMA k-steps)
#define LDA 72                // 64 + 8 pad halves (144 B rows, 16B aligned)
#define LDV 40                // 32 + 8 pad halves (80 B rows, 16B aligned)

typedef _Float16 h16;
typedef __attribute__((ext_vector_type(16))) _Float16 v16h;
typedef __attribute__((ext_vector_type(8)))  _Float16 v8h;
typedef __attribute__((ext_vector_type(4)))  _Float16 v4h;
typedef __attribute__((ext_vector_type(8)))  float    v8f;
typedef __attribute__((ext_vector_type(4)))  float    f32x4;

// ---------------------------------------------------------------- WMMA helpers
__device__ __forceinline__ v8f wmma_f16(v16h a, v16h b, v8f c) {
    return __builtin_amdgcn_wmma_f32_16x16x32_f16(
        false, a, false, b, (short)0, c, false, false);
}

// A-matrix 16x32 fragment, row-major source (row stride ld halves, rows 16B
// aligned). ISA layout makes the 16 halves two contiguous 8-half runs:
// lanes 0-15: M=lane, K={kh..kh+7, 16+kh..16+kh+7} with kh=0; lanes 16-31: kh=8.
__device__ __forceinline__ v16h load_a_frag(const h16* __restrict__ src, int ld) {
    const int lane = threadIdx.x & 31;
    const h16* p = src + (lane & 15) * ld + ((lane >> 4) * 8);
    const v8h lo = *(const v8h*)(p);
    const v8h hi = *(const v8h*)(p + 16);
    return __builtin_shufflevector(lo, hi, 0,1,2,3,4,5,6,7,8,9,10,11,12,13,14,15);
}

// B-matrix 32x16 fragment from N-major storage src[n][k] (contraction dim is
// the fast axis): lane n = lane&15, K = {kh..kh+7, 16+kh..16+kh+7}, kh = (lane>>4)*8.
__device__ __forceinline__ v16h load_bn_frag(const h16* __restrict__ src, int ld) {
    const int lane = threadIdx.x & 31;
    const h16* p = src + (lane & 15) * ld + ((lane >> 4) * 8);
    const v8h lo = *(const v8h*)(p);
    const v8h hi = *(const v8h*)(p + 16);
    return __builtin_shufflevector(lo, hi, 0,1,2,3,4,5,6,7,8,9,10,11,12,13,14,15);
}

// ---------------------------------------------------------------- kernel 1: QKV
// X(16384x1024) f32  x  [Wq|Wk|Wv](1024x1024) f32  -> f16 Q/K/V in [b][s][h][t][d]
__global__ __launch_bounds__(256) void qkv_gemm_kernel(
    const float* __restrict__ x,
    const float* __restrict__ Wq, const float* __restrict__ Wk, const float* __restrict__ Wv,
    const float* __restrict__ bq, const float* __restrict__ bk, const float* __restrict__ bv,
    h16* __restrict__ qb, h16* __restrict__ kb, h16* __restrict__ vb)
{
    __shared__ __align__(16) h16 As[TM][LDA];     // row-major tokens x k
    __shared__ __align__(16) h16 Bt[TN][LDA];     // N-major: Bt[n][k]

    const int tid   = threadIdx.x;
    const int row0  = blockIdx.x * TM;
    const int ncol0 = blockIdx.y * TN;            // 0..3071 across Wq|Wk|Wv
    const int which = ncol0 >> 10;                // uniform per block
    const int nc0   = ncol0 & (DIM - 1);

    const float* W    = (which == 0) ? Wq : (which == 1) ? Wk : Wv;
    const float* bias = (which == 0) ? bq : (which == 1) ? bk : bv;
    h16*         dst  = (which == 0) ? qb : (which == 1) ? kb : vb;

    const int w = tid >> 5, lane = tid & 31;
    const int wr = w & 3, wc = w >> 2;            // 4x2 wave grid, wave tile 32x64
    const int nloc = lane & 15, mbase = (lane >> 4) * 8;

    const v8f vzero = {};
    v8f acc[2][4];
#pragma unroll
    for (int mt = 0; mt < 2; ++mt)
#pragma unroll
        for (int nt = 0; nt < 4; ++nt) acc[mt][nt] = vzero;

    for (int k0 = 0; k0 < DIM; k0 += TKG) {
        // stage A: 128x64 f32 -> f16, vectorized (float4 in, 4-half out)
#pragma unroll
        for (int i = 0; i < (TM * TKG) / (4 * 256); ++i) {   // 8
            const int e = i * 256 + tid;
            const int r = e >> 4, c4 = e & 15;
            const f32x4 f = *(const f32x4*)&x[(size_t)(row0 + r) * DIM + k0 + c4 * 4];
            *(v4h*)&As[r][c4 * 4] = __builtin_convertvector(f, v4h);
        }
        // stage B transposed: Bt[n][k] = W[k0+k][nc0+n]
#pragma unroll
        for (int i = 0; i < (TKG * TN) / (4 * 256); ++i) {   // 8
            const int e = i * 256 + tid;
            const int r = e >> 5, c4 = e & 31;               // k row, 4-col group
            const f32x4 f = *(const f32x4*)&W[(size_t)(k0 + r) * DIM + nc0 + c4 * 4];
#pragma unroll
            for (int j = 0; j < 4; ++j) Bt[c4 * 4 + j][r] = (h16)f[j];
        }
        if (k0 + TKG < DIM)
            __builtin_prefetch(&x[(size_t)(row0 + (tid >> 1)) * DIM + k0 + TKG], 0, 1);
        __syncthreads();

#pragma unroll
        for (int ks = 0; ks < 2; ++ks) {
            v16h af0 = load_a_frag(&As[wr * 32][ks * 32],      LDA);
            v16h af1 = load_a_frag(&As[wr * 32 + 16][ks * 32], LDA);
#pragma unroll
            for (int nt = 0; nt < 4; ++nt) {
                v16h bf = load_bn_frag(&Bt[wc * 64 + nt * 16][ks * 32], LDA);
                acc[0][nt] = wmma_f16(af0, bf, acc[0][nt]);
                acc[1][nt] = wmma_f16(af1, bf, acc[1][nt]);
            }
        }
        __syncthreads();
    }

    // scatter-store into [b][s][h][t][d] f16 with bias
#pragma unroll
    for (int mt = 0; mt < 2; ++mt)
#pragma unroll
        for (int nt = 0; nt < 4; ++nt)
#pragma unroll
            for (int r = 0; r < 8; ++r) {
                const int gr = row0 + wr * 32 + mt * 16 + mbase + r;
                const int gn = nc0 + wc * 64 + nt * 16 + nloc;
                const int b = gr >> 13, l = gr & (LL - 1);
                const int s = l >> 9,   t = l & (SEG - 1);
                const int hh = gn >> 6, d = gn & (DK - 1);
                const size_t off = (((size_t)((b * SS + s) * HEADS + hh)) * SEG + t) * DK + d;
                dst[off] = (h16)(acc[mt][nt][r] + bias[gn]);
            }
}

// ---------------------------------------------------------------- kernel 2: attention
__global__ __launch_bounds__(256) void attn_kernel(
    const h16* __restrict__ qb, const h16* __restrict__ kb, const h16* __restrict__ vb,
    h16* __restrict__ ob)
{
    __shared__ __align__(16) h16   Ks[32][LDA];     // [key][d], rows 144B
    __shared__ __align__(16) h16   Vt[DK][LDV];     // transposed: [d][key]
    __shared__ __align__(16) float Srow[8][16][36];
    __shared__ __align__(16) h16   Ps[8][16][LDV];
    __shared__ float Arow[8][16];
    __shared__ float Mrow[8][16];
    __shared__ float Lrow[8][16];

    const int tid = threadIdx.x, w = tid >> 5, lane = tid & 31;
    const int nloc = lane & 15, mbase = (lane >> 4) * 8;

    const int bsh = blockIdx.x;
    const int b = bsh / (SS * HEADS);
    const int s = (bsh / HEADS) % SS;
    const int h = bsh % HEADS;
    const int q0 = blockIdx.y * 128 + w * 16;
    const float scale = 0.125f;  // 1/sqrt(DK)

    const size_t segbase = ((size_t)((b * SS + s) * HEADS + h)) * SEG * DK;
    const h16* qp = qb + segbase + (size_t)q0 * DK;
    const v16h qa0 = load_a_frag(qp, DK);
    const v16h qa1 = load_a_frag(qp + 32, DK);

    const v8f vzero = {};
    v8f fin[4];
#pragma unroll
    for (int nt = 0; nt < 4; ++nt) fin[nt] = vzero;

    // context list (uniform per block): self + valid dilated neighbors
    int   cseg[1 + 2 * NNE], cstep[1 + 2 * NNE], cnk[1 + 2 * NNE];
    float cwt[1 + 2 * NNE];
    int ncx = 0;
    cseg[ncx] = s; cstep[ncx] = 1; cnk[ncx] = SEG; cwt[ncx] = 1.0f; ++ncx;
#pragma unroll
    for (int off = 1; off <= NNE; ++off) {
        if (s - off >= 0) { cseg[ncx] = s - off; cstep[ncx] = DIL; cnk[ncx] = SEG / DIL; cwt[ncx] = 1.0f / (2 * NNE); ++ncx; }
        if (s + off < SS) { cseg[ncx] = s + off; cstep[ncx] = DIL; cnk[ncx] = SEG / DIL; cwt[ncx] = 1.0f / (2 * NNE); ++ncx; }
    }

    for (int ic = 0; ic < ncx; ++ic) {
        const size_t nbase = ((size_t)((b * SS + cseg[ic]) * HEADS + h)) * SEG * DK;
        const h16* kp = kb + nbase;
        const h16* vp = vb + nbase;
        const int step = cstep[ic], nk = cnk[ic];

        v8f o[4];
#pragma unroll
        for (int nt = 0; nt < 4; ++nt) o[nt] = vzero;
        if (lane < 16) { Mrow[w][lane] = -1e30f; Lrow[w][lane] = 0.0f; }

        for (int c0 = 0; c0 < nk; c0 += 32) {
            // stage 32 keys x 64 dims: K row-major (b128), V transposed
            {
                const int r = tid >> 3, c8 = (tid & 7) * 8;      // one v8h each
                const size_t goff = (size_t)((c0 + r) * step) * DK + c8;
                *(v8h*)&Ks[r][c8] = *(const v8h*)&kp[goff];
                const v8h vv = *(const v8h*)&vp[goff];
#pragma unroll
                for (int j = 0; j < 8; ++j) Vt[c8 + j][r] = vv[j];
            }
            __syncthreads();

            // scores: S(16x32) = Q(16x64) . K^T  (2 N-tiles x 2 k-steps)
#pragma unroll
            for (int nt = 0; nt < 2; ++nt) {
                v16h bk0 = load_bn_frag(&Ks[nt * 16][0],  LDA);
                v16h bk1 = load_bn_frag(&Ks[nt * 16][32], LDA);
                v8f sc = vzero;
                sc = wmma_f16(qa0, bk0, sc);
                sc = wmma_f16(qa1, bk1, sc);
#pragma unroll
                for (int r = 0; r < 8; ++r)
                    Srow[w][mbase + r][nt * 16 + nloc] = sc[r];
            }
            __syncthreads();

            // online softmax: lane L owns query row L of this wave's strip
            if (lane < 16) {
                const float mold = Mrow[w][lane];
                float mx = mold;
#pragma unroll
                for (int j = 0; j < 32; ++j) mx = fmaxf(mx, Srow[w][lane][j] * scale);
                const float alpha = __expf(mold - mx);
                float sum = 0.0f;
#pragma unroll
                for (int j = 0; j < 32; ++j) {
                    const float p = __expf(Srow[w][lane][j] * scale - mx);
                    Ps[w][lane][j] = (h16)p;
                    sum += p;
                }
                Mrow[w][lane] = mx;
                Lrow[w][lane] = Lrow[w][lane] * alpha + sum;
                Arow[w][lane] = alpha;
            }
            __syncthreads();

            // O = diag(alpha)*O + P(16x32) . V(32x64)
            const v16h pa = load_a_frag(&Ps[w][0][0], LDV);
#pragma unroll
            for (int nt = 0; nt < 4; ++nt) {
                v16h bv = load_bn_frag(&Vt[nt * 16][0], LDV);
#pragma unroll
                for (int r = 0; r < 8; ++r) o[nt][r] *= Arow[w][mbase + r];
                o[nt] = wmma_f16(pa, bv, o[nt]);
            }
            __syncthreads();
        }

        const float wgt = cwt[ic];
#pragma unroll
        for (int nt = 0; nt < 4; ++nt)
#pragma unroll
            for (int r = 0; r < 8; ++r)
                fin[nt][r] += wgt * o[nt][r] / Lrow[w][mbase + r];
    }

    // write f16 output in [b][l][channel] layout for the output projection
    const size_t orow0 = (size_t)(b * LL + s * SEG + q0);
#pragma unroll
    for (int nt = 0; nt < 4; ++nt)
#pragma unroll
        for (int r = 0; r < 8; ++r)
            ob[(orow0 + mbase + r) * DIM + h * DK + nt * 16 + nloc] = (h16)fin[nt][r];
}

// ---------------------------------------------------------------- kernel 3: out proj
__global__ __launch_bounds__(256) void out_gemm_kernel(
    const h16* __restrict__ Ain, const float* __restrict__ Wo,
    const float* __restrict__ bo, float* __restrict__ out)
{
    __shared__ __align__(16) h16 As[TM][LDA];
    __shared__ __align__(16) h16 Bt[TN][LDA];

    const int tid  = threadIdx.x;
    const int row0 = blockIdx.x * TM;
    const int nc0  = blockIdx.y * TN;

    const int w = tid >> 5, lane = tid & 31;
    const int wr = w & 3, wc = w >> 2;
    const int nloc = lane & 15, mbase = (lane >> 4) * 8;

    const v8f vzero = {};
    v8f acc[2][4];
#pragma unroll
    for (int mt = 0; mt < 2; ++mt)
#pragma unroll
        for (int nt = 0; nt < 4; ++nt) acc[mt][nt] = vzero;

    for (int k0 = 0; k0 < DIM; k0 += TKG) {
        // stage A (already f16): 128x64, one v8h per thread x4
#pragma unroll
        for (int i = 0; i < (TM * TKG) / (8 * 256); ++i) {   // 4
            const int e = i * 256 + tid;
            const int r = e >> 3, c8 = (e & 7) * 8;
            *(v8h*)&As[r][c8] = *(const v8h*)&Ain[(size_t)(row0 + r) * DIM + k0 + c8];
        }
        // stage B transposed: Bt[n][k] = Wo[k0+k][nc0+n]
#pragma unroll
        for (int i = 0; i < (TKG * TN) / (4 * 256); ++i) {   // 8
            const int e = i * 256 + tid;
            const int r = e >> 5, c4 = e & 31;
            const f32x4 f = *(const f32x4*)&Wo[(size_t)(k0 + r) * DIM + nc0 + c4 * 4];
#pragma unroll
            for (int j = 0; j < 4; ++j) Bt[c4 * 4 + j][r] = (h16)f[j];
        }
        __syncthreads();

#pragma unroll
        for (int ks = 0; ks < 2; ++ks) {
            v16h af0 = load_a_frag(&As[wr * 32][ks * 32],      LDA);
            v16h af1 = load_a_frag(&As[wr * 32 + 16][ks * 32], LDA);
#pragma unroll
            for (int nt = 0; nt < 4; ++nt) {
                v16h bf = load_bn_frag(&Bt[wc * 64 + nt * 16][ks * 32], LDA);
                acc[0][nt] = wmma_f16(af0, bf, acc[0][nt]);
                acc[1][nt] = wmma_f16(af1, bf, acc[1][nt]);
            }
        }
        __syncthreads();
    }

#pragma unroll
    for (int mt = 0; mt < 2; ++mt)
#pragma unroll
        for (int nt = 0; nt < 4; ++nt)
#pragma unroll
            for (int r = 0; r < 8; ++r) {
                const int gr = row0 + wr * 32 + mt * 16 + mbase + r;
                const int gn = nc0 + wc * 64 + nt * 16 + nloc;
                out[(size_t)gr * DIM + gn] = acc[mt][nt][r] + bo[gn];
            }
}

// ---------------------------------------------------------------- launcher
extern "C" void kernel_launch(void* const* d_in, const int* in_sizes, int n_in,
                              void* d_out, int out_size, void* d_ws, size_t ws_size,
                              hipStream_t stream) {
    (void)in_sizes; (void)n_in; (void)out_size; (void)ws_size;
    const float* x  = (const float*)d_in[0];
    const float* Wq = (const float*)d_in[1];
    const float* bq = (const float*)d_in[2];
    const float* Wk = (const float*)d_in[3];
    const float* bk = (const float*)d_in[4];
    const float* Wv = (const float*)d_in[5];
    const float* bv = (const float*)d_in[6];
    const float* Wo = (const float*)d_in[7];
    const float* bo = (const float*)d_in[8];
    float* out = (float*)d_out;

    const size_t nqkv = (size_t)BB * LL * DIM;     // elements per f16 buffer
    h16* qb = (h16*)d_ws;
    h16* kb = qb + nqkv;
    h16* vb = kb + nqkv;
    h16* ob = vb + nqkv;                           // total 128 MiB of workspace

    qkv_gemm_kernel<<<dim3(M_TOT / TM, (3 * DIM) / TN), 256, 0, stream>>>(
        x, Wq, Wk, Wv, bq, bk, bv, qb, kb, vb);
    attn_kernel<<<dim3(BB * SS * HEADS, SEG / 128), 256, 0, stream>>>(qb, kb, vb, ob);
    out_gemm_kernel<<<dim3(M_TOT / TM, DIM / TN), 256, 0, stream>>>(ob, Wo, bo, out);
}
